// SphereNetModel_63763084476730
// MI455X (gfx1250) — matrix operand order repaired and checked by reading, not compiled, not run on hardware
//
#include <hip/hip_runtime.h>

// ---------------- problem constants (match reference) ----------------
#define N_NODES 25000
#define N_EDGES 250000
#define N_TRIP  500000
#define N_GRAPH 128
#define HDIM    128
#define INTDIM  64
#define RDIM    6
#define LDIM    7
#define PI_F    3.14159265358979323846f

typedef __attribute__((ext_vector_type(16))) _Float16 v16h;
typedef __attribute__((ext_vector_type(8)))  _Float16 v8h;
typedef __attribute__((ext_vector_type(8)))  float    v8f;

__device__ __forceinline__ float swish_f(float x) { return x / (1.0f + __expf(-x)); }

// ---------------- f32 -> f16 transposed weight conversion ----------------
// src: [K,N] f32 row-major  ->  dst: [N,K] f16 row-major
__global__ void cvt_f16_t_kernel(const float* __restrict__ s, _Float16* __restrict__ d,
                                 int K, int N) {
  int t = blockIdx.x * blockDim.x + threadIdx.x;
  if (t >= K * N) return;
  int n = t / K, k = t - n * K;
  d[t] = (_Float16)s[k * N + n];
}

__global__ void zero_f32_kernel(float* __restrict__ p, long n) {
  long t = (long)blockIdx.x * blockDim.x + threadIdx.x;
  if (t < n) p[t] = 0.0f;
}

// ---------------- WMMA GEMM: Y(f16) = epilogue( A[M,KT] @ W[KT,N] ) ----------------
// block = 256 threads = 8 waves; wave w computes tile (blockIdx.x, w) of 16x16.
// A: f16 activations (AF32=false) or f32 atomic-accumulator buffer (AF32=true),
//    or gathered concat(h[i],h[j],rbf0) f16 when CONCAT.
// Wt: f16 TRANSPOSED [N,KT].  KT % 32 == 0, N % 16 == 0, N <= 128.
template<int KT, bool AF32, bool CONCAT, bool BIAS, bool ACT, bool MUL, bool RES>
__global__ void gemm_wmma(const void* __restrict__ A, const _Float16* __restrict__ Wt,
                          const float* __restrict__ bias, const _Float16* __restrict__ mulp,
                          const _Float16* __restrict__ resp, _Float16* __restrict__ Y,
                          int M, int Nc,
                          const _Float16* __restrict__ Hbuf, const int* __restrict__ gi,
                          const int* __restrict__ gj, const _Float16* __restrict__ R0) {
  int wave = threadIdx.x >> 5;
  int lane = threadIdx.x & 31;
  if (wave * 16 >= Nc) return;               // whole-wave exit: EXEC stays all-1 for WMMA
  int  mrow = lane & 15;
  int  hiA  = (lane >> 4) * 8;               // A k-base within a 32-chunk (0 or 8)
  int  hiB  = (lane >> 4) * 16;              // unused; B loads also use hiA (k-base 0 or 8)
  long row  = (long)blockIdx.x * 16 + mrow;
  long rowc = row < M ? row : (long)M - 1;   // clamp for safe loads on tail tile
  int  col  = wave * 16 + (lane & 15);
  const _Float16* wrow = Wt + (long)col * KT;
  const float*    arowf = nullptr;
  const _Float16* arowh = nullptr;
  long gI = 0, gJ = 0;
  if (CONCAT)     { gI = gi[rowc]; gJ = gj[rowc]; }
  else if (AF32)  arowf = (const float*)A + rowc * KT;
  else            arowh = (const _Float16*)A + rowc * KT;
  v8f acc = {};
#pragma unroll
  for (int k0 = 0; k0 < KT; k0 += 32) {
    v16h a;
    if (AF32) {
      // f32 source: 4x b128 loads + pack-convert
      const float* ab = arowf + k0;
      float4 a0 = *(const float4*)(ab + hiA);
      float4 a1 = *(const float4*)(ab + hiA + 4);
      float4 a2 = *(const float4*)(ab + hiA + 16);
      float4 a3 = *(const float4*)(ab + hiA + 20);
      a[0] = (_Float16)a0.x; a[1] = (_Float16)a0.y; a[2] = (_Float16)a0.z; a[3] = (_Float16)a0.w;
      a[4] = (_Float16)a1.x; a[5] = (_Float16)a1.y; a[6] = (_Float16)a1.z; a[7] = (_Float16)a1.w;
      a[8]  = (_Float16)a2.x; a[9]  = (_Float16)a2.y; a[10] = (_Float16)a2.z; a[11] = (_Float16)a2.w;
      a[12] = (_Float16)a3.x; a[13] = (_Float16)a3.y; a[14] = (_Float16)a3.z; a[15] = (_Float16)a3.w;
    } else {
      // f16 source: 2x b128 loads, no converts
      const _Float16* ab;
      if (CONCAT) {                          // k0 is compile-time after unroll
        if (k0 < 128)      ab = Hbuf + gI * 128 + k0;
        else if (k0 < 256) ab = Hbuf + gJ * 128 + (k0 - 128);
        else               ab = R0 + rowc * 128 + (k0 - 256);
      } else {
        ab = arowh + k0;
      }
      v8h x0 = *(const v8h*)(ab + hiA);
      v8h x1 = *(const v8h*)(ab + hiA + 16);
      a = __builtin_shufflevector(x0, x1, 0, 1, 2, 3, 4, 5, 6, 7,
                                  8, 9, 10, 11, 12, 13, 14, 15);
    }
    // ---- B fragment: 2x8 f16 along K of the transposed weight row (same k order as A) ----
    v8h b0 = *(const v8h*)(wrow + k0 + hiA);
    v8h b1 = *(const v8h*)(wrow + k0 + hiA + 16);
    v16h b = __builtin_shufflevector(b0, b1, 0, 1, 2, 3, 4, 5, 6, 7,
                                     8, 9, 10, 11, 12, 13, 14, 15);
    acc = __builtin_amdgcn_wmma_f32_16x16x32_f16(false, a, false, b, (short)0, acc, false, false);
  }
  int rbase = (lane < 16) ? 0 : 8;           // C/D layout: VGPR r -> M = r (+8 for hi lanes)
#pragma unroll
  for (int r = 0; r < 8; ++r) {
    long orow = (long)blockIdx.x * 16 + rbase + r;
    if (orow >= M) continue;
    float y = acc[r];
    if (BIAS) y += bias[col];
    if (ACT)  y = swish_f(y);
    if (MUL)  y *= (float)mulp[orow * Nc + col];
    if (RES)  y += (float)resp[orow * Nc + col];
    Y[orow * Nc + col] = (_Float16)y;
  }
}

// ---------------- spherical Bessel zeros/norms on device (double, 1 thread) ----------------
__device__ double jl_d(int l, double x) {
  double v0 = sin(x) / x;
  if (l == 0) return v0;
  double v1 = sin(x) / (x * x) - cos(x) / x;
  double a = v0, b = v1, c2 = v1;
  for (int ll = 2; ll <= l; ++ll) { c2 = (2.0 * ll - 1.0) / x * b - a; a = b; b = c2; }
  return (l == 1) ? v1 : c2;
}

__global__ void init_bessel_kernel(float* __restrict__ zeros, float* __restrict__ norms) {
  if (threadIdx.x != 0 || blockIdx.x != 0) return;
  const int NL = LDIM, NR = RDIM;
  double z[NL][NR + NL];
  for (int n = 0; n < NR + NL; ++n) z[0][n] = (n + 1) * 3.14159265358979323846;
  for (int l = 1; l < NL; ++l) {
    for (int n = 0; n < NR + NL - l; ++n) {
      double lo = z[l - 1][n], hi = z[l - 1][n + 1];
      double flo = jl_d(l, lo);
      for (int it = 0; it < 80; ++it) {
        double mid = 0.5 * (lo + hi);
        double fm = jl_d(l, mid);
        if (flo * fm <= 0.0) hi = mid; else { lo = mid; flo = fm; }
      }
      z[l][n] = 0.5 * (lo + hi);
    }
  }
  for (int l = 0; l < NL; ++l)
    for (int r = 0; r < NR; ++r) {
      zeros[l * NR + r] = (float)z[l][r];
      norms[l * NR + r] = (float)(sqrt(2.0 / 1000.0) / fabs(jl_d(l + 1, z[l][r])));
    }
}

// ---------------- per-edge geometry + radial bases ----------------
__device__ float sph_jl(int l, float x) {
  float xs = fmaxf(x, 0.5f);
  float v0 = sinf(xs) / xs;
  float v1 = sinf(xs) / (xs * xs) - cosf(xs) / xs;
  float rec = (l == 0) ? v0 : v1;
  float pm2 = v0, pm1 = v1;
  for (int ll = 2; ll <= l; ++ll) { float c2 = (2.0f * ll - 1.0f) / xs * pm1 - pm2; pm2 = pm1; pm1 = c2; rec = c2; }
  float dfact = 1.0f;
  for (int k = 1; k <= 2 * l + 1; k += 2) dfact *= (float)k;
  float t = 0.5f * x * x, term = 1.0f, ser = 1.0f;
  for (int k = 1; k <= 8; ++k) { term *= -t / (k * (2 * l + 2 * k + 1)); ser += term; }
  float xp = 1.0f;
  for (int q = 0; q < l; ++q) xp *= x;
  ser = xp / dfact * ser;
  return (x < 2.5f) ? ser : rec;
}

__global__ void edge_geom_kernel(const float* __restrict__ pos, const int* __restrict__ ei,
                                 const int* __restrict__ ej, const float* __restrict__ zeros,
                                 const float* __restrict__ norms, float* __restrict__ vec,
                                 float* __restrict__ dist, float* __restrict__ rbf,
                                 float* __restrict__ rlk) {
  long e = (long)blockIdx.x * blockDim.x + threadIdx.x;
  if (e >= N_EDGES) return;
  long a = ei[e], b = ej[e];
  float vx = pos[a * 3 + 0] - pos[b * 3 + 0];
  float vy = pos[a * 3 + 1] - pos[b * 3 + 1];
  float vz = pos[a * 3 + 2] - pos[b * 3 + 2];
  float dd = fmaxf(sqrtf(vx * vx + vy * vy + vz * vz + 1e-12f), 1e-2f);
  vec[e * 3 + 0] = vx; vec[e * 3 + 1] = vy; vec[e * 3 + 2] = vz;
  dist[e] = dd;
  float d = dd * 0.1f;                                  // dist / CUTOFF
  float x5 = d * d; x5 *= x5; x5 *= d;                  // d^5  (p-1 = 5)
  float env = 1.0f / d - 28.0f * x5 + 48.0f * x5 * d - 21.0f * x5 * d * d;
  for (int r = 0; r < RDIM; ++r) rbf[e * RDIM + r] = env * sinf((r + 1) * PI_F * d);
  for (int l = 0; l < LDIM; ++l)
    for (int r = 0; r < RDIM; ++r)
      rlk[e * 42 + l * RDIM + r] = norms[l * RDIM + r] * sph_jl(l, zeros[l * RDIM + r] * d);
}

// ---------------- per-triplet angular bases ----------------
__device__ __forceinline__ float sh_coef(int l, int am) {
  float fr = 1.0f;
  for (int k = l - am + 1; k <= l + am; ++k) fr *= (float)k;
  return sqrtf((2 * l + 1) / (4.0f * PI_F) / fr);
}

__global__ void triplet_basis_kernel(const float* __restrict__ vec, const float* __restrict__ dist,
                                     const float* __restrict__ rlk, const int* __restrict__ idx_kj,
                                     const int* __restrict__ idx_ji, float* __restrict__ sbf,
                                     float* __restrict__ rlkt, float* __restrict__ ylm) {
  long t = (long)blockIdx.x * blockDim.x + threadIdx.x;
  if (t >= N_TRIP) return;
  long ej = idx_ji[t], ek = idx_kj[t];
  float ax = vec[ej * 3 + 0], ay = vec[ej * 3 + 1], az = vec[ej * 3 + 2];
  float bx = vec[ek * 3 + 0], by = vec[ek * 3 + 1], bz = vec[ek * 3 + 2];
  float crx = ay * bz - az * by, cry = az * bx - ax * bz, crz = ax * by - ay * bx;
  float theta = atan2f(sqrtf(crx * crx + cry * cry + crz * crz + 1e-12f),
                       ax * bx + ay * by + az * bz);
  // p1 = v_ji x (1,0,0) = (0, az, -ay)
  float p1y = az, p1z = -ay;
  float qx = p1y * crz - p1z * cry;
  float qy = p1z * crx;
  float qz = -p1y * crx;
  float num = (qx * ax + qy * ay + qz * az) / dist[ej];
  float den = p1y * cry + p1z * crz;
  float phi = atan2f(num, den);
  float ct = cosf(theta), st = sinf(theta);
  float P[LDIM][LDIM];
  P[0][0] = 1.0f;
  for (int m = 1; m < LDIM; ++m) P[m][m] = -(2.0f * m - 1.0f) * st * P[m - 1][m - 1];
  for (int m = 0; m < LDIM - 1; ++m) P[m + 1][m] = (2.0f * m + 1.0f) * ct * P[m][m];
  for (int m = 0; m < LDIM; ++m)
    for (int l = m + 2; l < LDIM; ++l)
      P[l][m] = ((2.0f * l - 1.0f) * ct * P[l - 1][m] - (l + m - 1.0f) * P[l - 2][m]) / (float)(l - m);
  // cbf (m = 0) folded into sbf with gathered rlk
  for (int l = 0; l < LDIM; ++l) {
    float cl = sh_coef(l, 0) * P[l][0];
    for (int r = 0; r < RDIM; ++r) {
      float rv = rlk[ek * 42 + l * RDIM + r];
      rlkt[t * 42 + l * RDIM + r] = rv;
      sbf[t * 42 + l * RDIM + r] = cl * rv;
    }
  }
  // full real spherical harmonics -> 49 values (row-major [7,7] reshape)
  int idx = 0;
  for (int l = 0; l < LDIM; ++l)
    for (int m = -l; m <= l; ++m) {
      int am = m < 0 ? -m : m;
      float c = sh_coef(l, am);
      float v;
      if (m == 0)      v = c * P[l][0];
      else if (m > 0)  v = 1.41421356f * c * P[l][am] * cosf(am * phi);
      else             v = 1.41421356f * c * P[l][am] * sinf(am * phi);
      ylm[t * 49 + idx++] = v;
    }
}

// ---------------- small-K GEMMs (K=6 / 6->8->N chains), plain VALU, f16 out ----------------
template<bool BIAS, bool ACT, bool MUL>
__global__ void small6_kernel(const float* __restrict__ A, const float* __restrict__ W,
                              const float* __restrict__ bias, const _Float16* __restrict__ mulp,
                              _Float16* __restrict__ Y, long M, int Nc) {
  long t = (long)blockIdx.x * blockDim.x + threadIdx.x;
  if (t >= M * Nc) return;
  long m = t / Nc; int c = (int)(t - m * Nc);
  float s = 0.0f;
#pragma unroll
  for (int r = 0; r < RDIM; ++r) s += A[m * RDIM + r] * W[r * Nc + c];
  if (BIAS) s += bias[c];
  if (ACT)  s = swish_f(s);
  if (MUL)  s *= (float)mulp[t];
  Y[t] = (_Float16)s;
}

__global__ void small_chain_kernel(const float* __restrict__ A, const float* __restrict__ W1,
                                   const float* __restrict__ W2, _Float16* __restrict__ Y,
                                   long M, int Nc) {
  long t = (long)blockIdx.x * blockDim.x + threadIdx.x;
  if (t >= M * Nc) return;
  long m = t / Nc; int c = (int)(t - m * Nc);
  float a0[RDIM];
#pragma unroll
  for (int r = 0; r < RDIM; ++r) a0[r] = A[m * RDIM + r];
  float y = 0.0f;
#pragma unroll
  for (int b = 0; b < 8; ++b) {
    float s = 0.0f;
#pragma unroll
    for (int r = 0; r < RDIM; ++r) s += a0[r] * W1[r * 8 + b];
    y += s * W2[b * Nc + c];
  }
  Y[t] = (_Float16)y;
}

// ---------------- per-layer triplet gate + segment-sum over idx_ji ----------------
__global__ void triplet_layer_kernel(const float* __restrict__ sbf, const float* __restrict__ rlkt,
                                     const float* __restrict__ ylm, const _Float16* __restrict__ xdown,
                                     const int* __restrict__ idx_kj, const int* __restrict__ idx_ji,
                                     const float* __restrict__ w_sbf1, const float* __restrict__ w_sbf2,
                                     const float* __restrict__ w_t1, const float* __restrict__ w_t2,
                                     float* __restrict__ acc) {
  __shared__ float s_w1[42 * 8], s_w2[8 * 64], s_t1[294 * 8], s_t2[8 * 64];
  for (int q = threadIdx.x; q < 42 * 8; q += blockDim.x) s_w1[q] = w_sbf1[q];
  for (int q = threadIdx.x; q < 8 * 64; q += blockDim.x) { s_w2[q] = w_sbf2[q]; s_t2[q] = w_t2[q]; }
  for (int q = threadIdx.x; q < 294 * 8; q += blockDim.x) s_t1[q] = w_t1[q];
  __syncthreads();
  long t = (long)blockIdx.x * blockDim.x + threadIdx.x;
  if (t >= N_TRIP) return;
  float s8[8] = {0, 0, 0, 0, 0, 0, 0, 0};
  float t8[8] = {0, 0, 0, 0, 0, 0, 0, 0};
  const float* sb = sbf + t * 42;
  for (int q = 0; q < 42; ++q) {
    float v = sb[q];
#pragma unroll
    for (int b = 0; b < 8; ++b) s8[b] += v * s_w1[q * 8 + b];
  }
  const float* rl = rlkt + t * 42;
  const float* yl = ylm + t * 49;
  for (int a = 0; a < LDIM; ++a)
    for (int l = 0; l < LDIM; ++l) {
      float yv = yl[a * LDIM + l];
      for (int r = 0; r < RDIM; ++r) {
        float c2 = yv * rl[l * RDIM + r];
        int rowi = (a * 42 + l * RDIM + r) * 8;
#pragma unroll
        for (int b = 0; b < 8; ++b) t8[b] += c2 * s_t1[rowi + b];
      }
    }
  long ek = idx_kj[t], ej = idx_ji[t];
  const _Float16* xd = xdown + ek * INTDIM;
  float* dst = acc + ej * INTDIM;
  for (int c = 0; c < INTDIM; ++c) {
    float sp = 0.0f, tp = 0.0f;
#pragma unroll
    for (int b = 0; b < 8; ++b) { sp += s8[b] * s_w2[b * INTDIM + c]; tp += t8[b] * s_t2[b * INTDIM + c]; }
    atomicAdd(dst + c, (float)xd[c] * sp * tp);
  }
}

// ---------------- gathers / scatters / output reduction ----------------
__global__ void gather_h_kernel(const float* __restrict__ emb, const int* __restrict__ z,
                                _Float16* __restrict__ h) {
  long t = (long)blockIdx.x * blockDim.x + threadIdx.x;
  if (t >= (long)N_NODES * HDIM) return;
  long n = t >> 7; int c = (int)(t & 127);
  h[t] = (_Float16)emb[(long)z[n] * HDIM + c];
}

__global__ void scatter_add128_kernel(const _Float16* __restrict__ src, const int* __restrict__ seg,
                                      float* __restrict__ dst, long rows) {
  long t = (long)blockIdx.x * blockDim.x + threadIdx.x;
  if (t >= rows * HDIM) return;
  long r = t >> 7; int c = (int)(t & 127);
  atomicAdd(&dst[(long)seg[r] * HDIM + c], (float)src[t]);
}

__global__ void node_out_kernel(const _Float16* __restrict__ v, const float* __restrict__ w,
                                const int* __restrict__ batch, float* __restrict__ out) {
  long t = (long)blockIdx.x * blockDim.x + threadIdx.x;
  if (t >= N_NODES) return;
  float s = 0.0f;
  for (int c = 0; c < HDIM; ++c) s += (float)v[t * HDIM + c] * w[c];
  atomicAdd(&out[batch[t]], s);
}

// ---------------- host-side orchestration ----------------
struct ELayer {
  const float *w_ji, *b_ji, *w_kj, *b_kj, *w_rbf1, *w_rbf2, *w_sbf1, *w_sbf2, *w_t1, *w_t2;
  const float *w_down, *w_up, *bw1, *bb1, *bw2, *bb2, *w_mid, *b_mid;
  const float *a1w1, *a1b1, *a1w2, *a1b2, *a2w1, *a2b1, *a2w2, *a2b2, *w_out;
};
struct VLayer { const float *w_up, *l0w, *l0b, *l1w, *l1b, *w_out; };

static inline unsigned g1d(long n) { return (unsigned)((n + 255) / 256); }

extern "C" void kernel_launch(void* const* d_in, const int* in_sizes, int n_in,
                              void* d_out, int out_size, void* d_ws, size_t ws_size,
                              hipStream_t stream) {
  (void)in_sizes; (void)n_in; (void)out_size; (void)ws_size;
  int cur = 0;
  auto nf = [&]() { return (const float*)d_in[cur++]; };
  auto ni = [&]() { return (const int*)d_in[cur++]; };

  // ---- params (setup_inputs insertion order) ----
  const float* node_emb = nf();
  const float* iw_rbf0 = nf(); const float* ib_rbf0 = nf();
  const float* iw_cat = nf();  const float* ib_cat = nf();
  const float* iw_rbf1 = nf();
  ELayer el[4];
  for (int l = 0; l < 4; ++l) {
    ELayer& e = el[l];
    e.w_ji = nf(); e.b_ji = nf(); e.w_kj = nf(); e.b_kj = nf();
    e.w_rbf1 = nf(); e.w_rbf2 = nf(); e.w_sbf1 = nf(); e.w_sbf2 = nf();
    e.w_t1 = nf(); e.w_t2 = nf(); e.w_down = nf(); e.w_up = nf();
    e.bw1 = nf(); e.bb1 = nf(); e.bw2 = nf(); e.bb2 = nf();
    e.w_mid = nf(); e.b_mid = nf();
    e.a1w1 = nf(); e.a1b1 = nf(); e.a1w2 = nf(); e.a1b2 = nf();
    e.a2w1 = nf(); e.a2b1 = nf(); e.a2w2 = nf(); e.a2b2 = nf();
    e.w_out = nf();
  }
  VLayer vl[5];
  for (int l = 0; l < 5; ++l) {
    vl[l].w_up = nf(); vl[l].l0w = nf(); vl[l].l0b = nf();
    vl[l].l1w = nf(); vl[l].l1b = nf(); vl[l].w_out = nf();
  }
  const float* pos = nf();
  const int* z = ni(); const int* batch = ni();
  const int* ei = ni(); const int* ej = ni();
  const int* idx_kj = ni(); const int* idx_ji = ni();

  // ---- workspace layout ----
  char* ws = (char*)d_ws;
  size_t wcur = 0;
  auto alloc = [&](size_t bytes) -> void* {
    void* p = ws + wcur;
    wcur += (bytes + 255) & ~(size_t)255;
    return p;
  };
  float* zeros = (float*)alloc(42 * 4);
  float* norms = (float*)alloc(42 * 4);
  float* vacc  = (float*)alloc((size_t)N_NODES * HDIM * 4);   // atomic accumulator (f32)
  float* vec   = (float*)alloc((size_t)N_EDGES * 3 * 4);
  float* dist  = (float*)alloc((size_t)N_EDGES * 4);
  float* rbf   = (float*)alloc((size_t)N_EDGES * RDIM * 4);
  float* rlk   = (float*)alloc((size_t)N_EDGES * 42 * 4);
  float* sbf   = (float*)alloc((size_t)N_TRIP * 42 * 4);
  float* rlkt  = (float*)alloc((size_t)N_TRIP * 42 * 4);
  float* ylm   = (float*)alloc((size_t)N_TRIP * 49 * 4);
  float* Bacc  = (float*)alloc((size_t)N_EDGES * INTDIM * 4); // atomic accumulator (f32)
  _Float16* h    = (_Float16*)alloc((size_t)N_NODES * HDIM * 2);
  _Float16* v1   = (_Float16*)alloc((size_t)N_NODES * HDIM * 2);
  _Float16* v2   = (_Float16*)alloc((size_t)N_NODES * HDIM * 2);
  _Float16* Be1  = (_Float16*)alloc((size_t)N_EDGES * HDIM * 2);
  _Float16* Bxji = (_Float16*)alloc((size_t)N_EDGES * HDIM * 2);
  _Float16* Bxkj = (_Float16*)alloc((size_t)N_EDGES * HDIM * 2);
  _Float16* Bg   = (_Float16*)alloc((size_t)N_EDGES * HDIM * 2);
  _Float16* Bxd  = (_Float16*)alloc((size_t)N_EDGES * INTDIM * 2);

  // ---- convert big weights to transposed f16 (sequential on stream -> ordered) ----
  auto half_t = [&](const float* src, int K, int N) -> _Float16* {
    _Float16* dst = (_Float16*)alloc((size_t)K * N * 2);
    cvt_f16_t_kernel<<<g1d((long)K * N), 256, 0, stream>>>(src, dst, K, N);
    return dst;
  };
  _Float16* hw_cat = half_t(iw_cat, 384, 128);
  _Float16 *hw_ji[4], *hw_kj[4], *hw_down[4], *hw_up[4], *hw_b1[4], *hw_b2[4], *hw_mid[4];
  _Float16 *hw_a1w1[4], *hw_a1w2[4], *hw_a2w1[4], *hw_a2w2[4];
  for (int l = 0; l < 4; ++l) {
    hw_ji[l]   = half_t(el[l].w_ji, 128, 128);
    hw_kj[l]   = half_t(el[l].w_kj, 128, 128);
    hw_down[l] = half_t(el[l].w_down, 128, 64);
    hw_up[l]   = half_t(el[l].w_up, 64, 128);
    hw_b1[l]   = half_t(el[l].bw1, 128, 128);
    hw_b2[l]   = half_t(el[l].bw2, 128, 128);
    hw_mid[l]  = half_t(el[l].w_mid, 128, 128);
    hw_a1w1[l] = half_t(el[l].a1w1, 128, 128);
    hw_a1w2[l] = half_t(el[l].a1w2, 128, 128);
    hw_a2w1[l] = half_t(el[l].a2w1, 128, 128);
    hw_a2w2[l] = half_t(el[l].a2w2, 128, 128);
  }
  _Float16* hv_up = half_t(vl[4].w_up, 128, 128);
  _Float16* hv_l0 = half_t(vl[4].l0w, 128, 128);
  _Float16* hv_l1 = half_t(vl[4].l1w, 128, 128);

  // ---- geometry & bases ----
  init_bessel_kernel<<<1, 32, 0, stream>>>(zeros, norms);
  gather_h_kernel<<<g1d((long)N_NODES * HDIM), 256, 0, stream>>>(node_emb, z, h);
  edge_geom_kernel<<<g1d(N_EDGES), 256, 0, stream>>>(pos, ei, ej, zeros, norms, vec, dist, rbf, rlk);
  triplet_basis_kernel<<<g1d(N_TRIP), 256, 0, stream>>>(vec, dist, rlk, idx_kj, idx_ji, sbf, rlkt, ylm);

  dim3 tb(256);
  dim3 gbE((unsigned)((N_EDGES + 15) / 16));
  dim3 gbN((unsigned)((N_NODES + 15) / 16));

  // ---- init block: rbf0 = swish(rbf @ w_rbf0 + b);  e1 = swish(cat @ w_cat + b) ----
  small6_kernel<true, true, false><<<g1d((long)N_EDGES * HDIM), 256, 0, stream>>>(
      rbf, iw_rbf0, ib_rbf0, nullptr, Bg, N_EDGES, HDIM);
  gemm_wmma<384, false, true, true, true, false, false><<<gbE, tb, 0, stream>>>(
      nullptr, hw_cat, ib_cat, nullptr, nullptr, Be1, N_EDGES, HDIM, h, ei, ej, Bg);
  (void)iw_rbf1;  // init e2 only feeds a dead _update_v (v is overwritten each layer)

  // ---- 4 interaction layers ----
  for (int l = 0; l < 4; ++l) {
    const ELayer& e = el[l];
    // x_ji = swish(x1 @ w_ji + b)
    gemm_wmma<128, false, false, true, true, false, false><<<gbE, tb, 0, stream>>>(
        Be1, hw_ji[l], e.b_ji, nullptr, nullptr, Bxji, N_EDGES, 128,
        nullptr, nullptr, nullptr, nullptr);
    // g = (rbf @ w_rbf1) @ w_rbf2
    small_chain_kernel<<<g1d((long)N_EDGES * HDIM), 256, 0, stream>>>(
        rbf, e.w_rbf1, e.w_rbf2, Bg, N_EDGES, HDIM);
    // x_kj = swish(x1 @ w_kj + b) * g
    gemm_wmma<128, false, false, true, true, true, false><<<gbE, tb, 0, stream>>>(
        Be1, hw_kj[l], e.b_kj, Bg, nullptr, Bxkj, N_EDGES, 128,
        nullptr, nullptr, nullptr, nullptr);
    // x_down = swish(x_kj @ w_down)   [E,64]
    gemm_wmma<128, false, false, false, true, false, false><<<gbE, tb, 0, stream>>>(
        Bxkj, hw_down[l], nullptr, nullptr, nullptr, Bxd, N_EDGES, 64,
        nullptr, nullptr, nullptr, nullptr);
    // triplet gate + segment_sum(idx_ji)
    zero_f32_kernel<<<g1d((long)N_EDGES * INTDIM), 256, 0, stream>>>(Bacc, (long)N_EDGES * INTDIM);
    triplet_layer_kernel<<<g1d(N_TRIP), 256, 0, stream>>>(
        sbf, rlkt, ylm, Bxd, idx_kj, idx_ji, e.w_sbf1, e.w_sbf2, e.w_t1, e.w_t2, Bacc);
    // e1' = x_ji + swish(acc @ w_up)   (f32 A; in place on Bxji)
    gemm_wmma<64, true, false, false, true, false, true><<<gbE, tb, 0, stream>>>(
        Bacc, hw_up[l], nullptr, nullptr, Bxji, Bxji, N_EDGES, 128,
        nullptr, nullptr, nullptr, nullptr);
    // before residual: cur += swish(swish(cur@w1+b1)@w2+b2)
    gemm_wmma<128, false, false, true, true, false, false><<<gbE, tb, 0, stream>>>(
        Bxji, hw_b1[l], e.bb1, nullptr, nullptr, Bg, N_EDGES, 128,
        nullptr, nullptr, nullptr, nullptr);
    gemm_wmma<128, false, false, true, true, false, true><<<gbE, tb, 0, stream>>>(
        Bg, hw_b2[l], e.bb2, nullptr, Bxji, Bxji, N_EDGES, 128,
        nullptr, nullptr, nullptr, nullptr);
    // e1 = swish(cur @ w_mid + b) + x1   (x1 = old Be1; write Be1 in place)
    gemm_wmma<128, false, false, true, true, false, true><<<gbE, tb, 0, stream>>>(
        Bxji, hw_mid[l], e.b_mid, nullptr, Be1, Be1, N_EDGES, 128,
        nullptr, nullptr, nullptr, nullptr);
    // after residual x2
    gemm_wmma<128, false, false, true, true, false, false><<<gbE, tb, 0, stream>>>(
        Be1, hw_a1w1[l], e.a1b1, nullptr, nullptr, Bg, N_EDGES, 128,
        nullptr, nullptr, nullptr, nullptr);
    gemm_wmma<128, false, false, true, true, false, true><<<gbE, tb, 0, stream>>>(
        Bg, hw_a1w2[l], e.a1b2, nullptr, Be1, Be1, N_EDGES, 128,
        nullptr, nullptr, nullptr, nullptr);
    gemm_wmma<128, false, false, true, true, false, false><<<gbE, tb, 0, stream>>>(
        Be1, hw_a2w1[l], e.a2b1, nullptr, nullptr, Bg, N_EDGES, 128,
        nullptr, nullptr, nullptr, nullptr);
    gemm_wmma<128, false, false, true, true, false, true><<<gbE, tb, 0, stream>>>(
        Bg, hw_a2w2[l], e.a2b2, nullptr, Be1, Be1, N_EDGES, 128,
        nullptr, nullptr, nullptr, nullptr);
  }

  // ---- e2 = (rbf @ w_out) * e1 (last layer only; earlier e2 feeds dead code) ----
  small6_kernel<false, false, true><<<g1d((long)N_EDGES * HDIM), 256, 0, stream>>>(
      rbf, el[3].w_out, nullptr, Be1, Bg, N_EDGES, HDIM);

  // ---- final update_v (params['v'][4]) + graph readout ----
  zero_f32_kernel<<<g1d((long)N_NODES * HDIM), 256, 0, stream>>>(vacc, (long)N_NODES * HDIM);
  scatter_add128_kernel<<<g1d((long)N_EDGES * HDIM), 256, 0, stream>>>(Bg, ei, vacc, N_EDGES);
  gemm_wmma<128, true, false, false, true, false, false><<<gbN, tb, 0, stream>>>(
      vacc, hv_up, nullptr, nullptr, nullptr, v1, N_NODES, 128,
      nullptr, nullptr, nullptr, nullptr);
  gemm_wmma<128, false, false, true, true, false, false><<<gbN, tb, 0, stream>>>(
      v1, hv_l0, vl[4].l0b, nullptr, nullptr, v2, N_NODES, 128,
      nullptr, nullptr, nullptr, nullptr);
  gemm_wmma<128, false, false, true, true, false, false><<<gbN, tb, 0, stream>>>(
      v2, hv_l1, vl[4].l1b, nullptr, nullptr, v1, N_NODES, 128,
      nullptr, nullptr, nullptr, nullptr);
  zero_f32_kernel<<<1, 256, 0, stream>>>((float*)d_out, N_GRAPH);
  node_out_kernel<<<g1d(N_NODES), 256, 0, stream>>>(v1, vl[4].w_out, batch, (float*)d_out);
}
